// FlashLlamaAttention_4320737100116
// MI455X (gfx1250) — matrix operand order repaired and checked by reading, not compiled, not run on hardware
//
#include <hip/hip_runtime.h>

// ---------------------------------------------------------------------------
// Types for CDNA5 WMMA (wave32): 16x16x32 bf16 -> f32
// ---------------------------------------------------------------------------
typedef __bf16 bf16;
typedef __attribute__((ext_vector_type(16))) __bf16 bf16x16;
typedef __attribute__((ext_vector_type(8)))  __bf16 bf16x8;
typedef __attribute__((ext_vector_type(8)))  float  f32x8;

// A-fragment: per lane one contiguous 16-elem K run (k = 16*half + s), as lo/hi 8-elem halves
union AF { bf16x16 v; struct { bf16x8 lo, hi; } p; };

__device__ __forceinline__ f32x8 wmma_bf16(bf16x16 a, bf16x16 b, f32x8 c) {
  return __builtin_amdgcn_wmma_f32_16x16x32_bf16(false, a, false, b, (short)0, c,
                                                 false, false);
}

__device__ __forceinline__ f32x8 zero8() {
  f32x8 z = {0.f, 0.f, 0.f, 0.f, 0.f, 0.f, 0.f, 0.f};
  return z;
}

// Problem constants
#define TT   4096   // B*S tokens
#define HID  4096
#define NQKV 6144
#define NH   32
#define NKVH 8
#define DD   128
#define SS   1024
#define NSLOTS 8192
#define SCALE 0.08838834764831845f  // 128^-0.5

// ---------------------------------------------------------------------------
// Elementwise helpers
// ---------------------------------------------------------------------------
__global__ void cvt_bf16_kernel(const float* __restrict__ in,
                                bf16* __restrict__ out, long n) {
  long i = (long)blockIdx.x * blockDim.x + threadIdx.x;
  long stride = (long)gridDim.x * blockDim.x;
  for (; i < n; i += stride) out[i] = (bf16)in[i];
}

__global__ void copy_f32_kernel(const float* __restrict__ in,
                                float* __restrict__ out, long n) {
  long i = (long)blockIdx.x * blockDim.x + threadIdx.x;
  long stride = (long)gridDim.x * blockDim.x;
  for (; i < n; i += stride) out[i] = in[i];
}

// ---------------------------------------------------------------------------
// GEMM: C[M,N] (f32) = A[M,K] (bf16 row-major) x B[N,K]^T (bf16 row-major)
// block = 128 threads = 4 waves; each wave -> 32(M) x 64(N) tile
// (each B fragment is reused by two A fragments -> 2x FLOP/byte on B stream)
// ---------------------------------------------------------------------------
__global__ __launch_bounds__(128)
void gemm_bf16_tn(const bf16* __restrict__ A, const bf16* __restrict__ Bm,
                  float* __restrict__ C, int M, int N, int K) {
  const int lane = threadIdx.x & 31;
  const int wave = threadIdx.x >> 5;
  const int l16  = lane & 15;
  const int half = lane >> 4;
  const int m0 = blockIdx.y * 128 + wave * 32;  // 2 row-fragments per wave
  const int n0 = blockIdx.x * 64;

  f32x8 acc[2][4];
#pragma unroll
  for (int u = 0; u < 2; ++u)
#pragma unroll
    for (int t = 0; t < 4; ++t) acc[u][t] = zero8();

  const bf16* arow0 = A + (size_t)(m0 + l16) * K;
  const bf16* arow1 = A + (size_t)(m0 + 16 + l16) * K;
  for (int k0 = 0; k0 < K; k0 += 32) {
    __builtin_prefetch(arow0 + k0 + 256, 0, 1);  // global_prefetch_b8
    AF a0, a1;
    a0.p.lo = *(const bf16x8*)(arow0 + k0 + 16 * half);
    a0.p.hi = *(const bf16x8*)(arow0 + k0 + 8 + 16 * half);
    a1.p.lo = *(const bf16x8*)(arow1 + k0 + 16 * half);
    a1.p.hi = *(const bf16x8*)(arow1 + k0 + 8 + 16 * half);
#pragma unroll
    for (int t = 0; t < 4; ++t) {
      const bf16* brow = Bm + (size_t)(n0 + t * 16 + l16) * K + k0 + 16 * half;
      bf16x16 b = *(const bf16x16*)brow;
      acc[0][t] = wmma_bf16(a0.v, b, acc[0][t]);
      acc[1][t] = wmma_bf16(a1.v, b, acc[1][t]);
    }
  }
  // C layout (ISA 7.12.2): lane -> n = lane%16, m = vgpr + 8*(lane/16)
#pragma unroll
  for (int u = 0; u < 2; ++u) {
#pragma unroll
    for (int t = 0; t < 4; ++t) {
#pragma unroll
      for (int r = 0; r < 8; ++r) {
        C[(size_t)(m0 + u * 16 + r + 8 * half) * N + (n0 + t * 16 + l16)] =
            acc[u][t][r];
      }
    }
  }
}

// ---------------------------------------------------------------------------
// RoPE + KV scatter: grid (T, H), block 64 (= D/2)
//  - writes roped q (bf16), roped k (bf16), V^T (bf16, [b,kvh,D,S])
//  - writes roped k and raw v (f32) into output caches at slots[t]
// ---------------------------------------------------------------------------
__global__ __launch_bounds__(64)
void rope_scatter_kernel(const float* __restrict__ qkv,
                         const float* __restrict__ cosp,
                         const float* __restrict__ sinp,
                         const int* __restrict__ slots,
                         bf16* __restrict__ qbf, bf16* __restrict__ kbf,
                         bf16* __restrict__ vT,
                         float* __restrict__ kc_out,
                         float* __restrict__ vc_out) {
  const int t = blockIdx.x;
  const int h = blockIdx.y;
  const int d = threadIdx.x;  // 0..63
  const float c = cosp[(size_t)t * 64 + d];
  const float s = sinp[(size_t)t * 64 + d];
  const float* row = qkv + (size_t)t * NQKV;

  // q
  {
    const float x1 = row[h * DD + d];
    const float x2 = row[h * DD + 64 + d];
    const size_t qo = ((size_t)t * NH + h) * DD;
    qbf[qo + d]      = (bf16)(x1 * c - x2 * s);
    qbf[qo + 64 + d] = (bf16)(x1 * s + x2 * c);
  }
  if (h < NKVH) {
    const int slot = slots[t];
    const size_t co = ((size_t)slot * NKVH + h) * DD;
    // k (roped)
    const float k1 = row[NH * DD + h * DD + d];
    const float k2 = row[NH * DD + h * DD + 64 + d];
    const float kr1 = k1 * c - k2 * s;
    const float kr2 = k1 * s + k2 * c;
    const size_t ko = ((size_t)t * NKVH + h) * DD;
    kbf[ko + d]      = (bf16)kr1;
    kbf[ko + 64 + d] = (bf16)kr2;
    kc_out[co + d]      = kr1;
    kc_out[co + 64 + d] = kr2;
    // v (raw) -> cache + transposed bf16 for PV wmma
    const float v1 = row[NH * DD + NKVH * DD + h * DD + d];
    const float v2 = row[NH * DD + NKVH * DD + h * DD + 64 + d];
    vc_out[co + d]      = v1;
    vc_out[co + 64 + d] = v2;
    const int b  = t >> 10;     // t / S
    const int si = t & 1023;    // t % S
    const size_t vo = (size_t)(b * NKVH + h) * DD * SS;
    vT[vo + (size_t)d * SS + si]        = (bf16)v1;
    vT[vo + (size_t)(64 + d) * SS + si] = (bf16)v2;
  }
}

// ---------------------------------------------------------------------------
// Flash attention: grid (S/16, B*H), block = 32 (one wave).
// Each wave: 16 query rows, loop key blocks of 32, online softmax.
// ---------------------------------------------------------------------------
__global__ __launch_bounds__(32)
void flash_attn_kernel(const bf16* __restrict__ qbf, const bf16* __restrict__ kbf,
                       const bf16* __restrict__ vT, bf16* __restrict__ obf) {
  __shared__ bf16 lds_p[16 * 32];  // P tile, row-major 16x32

  const int lane = threadIdx.x & 31;
  const int l16  = lane & 15;
  const int half = lane >> 4;
  const int q0 = blockIdx.x * 16;
  const int bh = blockIdx.y;
  const int b  = bh >> 5;   // / NH
  const int h  = bh & 31;   // % NH
  const int hk = h >> 2;    // / (NH/NKVH)

  // Q fragments: 4 K-chunks of 32 over D=128
  AF qf[4];
  const bf16* qrow = qbf + ((size_t)(b * SS + q0 + l16) * NH + h) * DD;
#pragma unroll
  for (int kc = 0; kc < 4; ++kc) {
    qf[kc].p.lo = *(const bf16x8*)(qrow + kc * 32 + 16 * half);
    qf[kc].p.hi = *(const bf16x8*)(qrow + kc * 32 + 8 + 16 * half);
  }

  f32x8 o[8];
#pragma unroll
  for (int nt = 0; nt < 8; ++nt) o[nt] = zero8();
  float rmax[8], rsum[8];
#pragma unroll
  for (int r = 0; r < 8; ++r) { rmax[r] = -1e30f; rsum[r] = 0.f; }

  const bf16* vbase = vT + (size_t)(b * NKVH + hk) * DD * SS;

  for (int j0 = 0; j0 <= q0 + 15; j0 += 32) {
    // scores S = Q K^T for keys [j0, j0+32)
    f32x8 s0 = zero8(), s1 = zero8();
#pragma unroll
    for (int kc = 0; kc < 4; ++kc) {
      const bf16* kp0 = kbf + ((size_t)(b * SS + j0 + l16) * NKVH + hk) * DD +
                        kc * 32 + 16 * half;
      const bf16* kp1 = kbf + ((size_t)(b * SS + j0 + 16 + l16) * NKVH + hk) * DD +
                        kc * 32 + 16 * half;
      s0 = wmma_bf16(qf[kc].v, *(const bf16x16*)kp0, s0);
      s1 = wmma_bf16(qf[kc].v, *(const bf16x16*)kp1, s1);
    }
    // scale + causal mask + online softmax (rows live in 16-lane halves)
#pragma unroll
    for (int r = 0; r < 8; ++r) {
      const int q = q0 + r + 8 * half;
      float va = s0[r] * SCALE;
      float vb = s1[r] * SCALE;
      if (j0 + l16 > q)      va = -1e30f;
      if (j0 + 16 + l16 > q) vb = -1e30f;
      float mx = fmaxf(va, vb);
      mx = fmaxf(mx, __shfl_xor(mx, 1, 16));
      mx = fmaxf(mx, __shfl_xor(mx, 2, 16));
      mx = fmaxf(mx, __shfl_xor(mx, 4, 16));
      mx = fmaxf(mx, __shfl_xor(mx, 8, 16));
      const float nm = fmaxf(rmax[r], mx);
      const float corr = __expf(rmax[r] - nm);
      rmax[r] = nm;
      const float pa = __expf(va - nm);
      const float pb = __expf(vb - nm);
      float rs = pa + pb;
      rs += __shfl_xor(rs, 1, 16);
      rs += __shfl_xor(rs, 2, 16);
      rs += __shfl_xor(rs, 4, 16);
      rs += __shfl_xor(rs, 8, 16);
      rsum[r] = rsum[r] * corr + rs;
#pragma unroll
      for (int nt = 0; nt < 8; ++nt) o[nt][r] *= corr;
      const int m = r + 8 * half;
      lds_p[m * 32 + l16]      = (bf16)pa;
      lds_p[m * 32 + 16 + l16] = (bf16)pb;
    }
    __syncthreads();
    // P as A-fragment (16x32), V^T tiles as B-fragments
    AF pf;
    pf.p.lo = *(const bf16x8*)(&lds_p[l16 * 32 + 16 * half]);
    pf.p.hi = *(const bf16x8*)(&lds_p[l16 * 32 + 8 + 16 * half]);
#pragma unroll
    for (int nt = 0; nt < 8; ++nt) {
      const bf16* vp = vbase + (size_t)(nt * 16 + l16) * SS + j0 + 16 * half;
      o[nt] = wmma_bf16(pf.v, *(const bf16x16*)vp, o[nt]);
    }
    __syncthreads();
  }

  // normalize and store as bf16 [T, H*D] for the output-projection GEMM
#pragma unroll
  for (int nt = 0; nt < 8; ++nt) {
#pragma unroll
    for (int r = 0; r < 8; ++r) {
      const int m = r + 8 * half;
      obf[(size_t)(b * SS + q0 + m) * (NH * DD) + h * DD + nt * 16 + l16] =
          (bf16)(o[nt][r] / rsum[r]);
    }
  }
}

// ---------------------------------------------------------------------------
// Launch
// ---------------------------------------------------------------------------
extern "C" void kernel_launch(void* const* d_in, const int* in_sizes, int n_in,
                              void* d_out, int out_size, void* d_ws, size_t ws_size,
                              hipStream_t stream) {
  const float* hidden    = (const float*)d_in[0];
  const float* cosp      = (const float*)d_in[1];
  const float* sinp      = (const float*)d_in[2];
  const float* wqkv      = (const float*)d_in[3];
  const float* wo        = (const float*)d_in[4];
  const float* kcache_in = (const float*)d_in[5];
  const float* vcache_in = (const float*)d_in[6];
  const int*   slots     = (const int*)d_in[7];

  float* out    = (float*)d_out;
  float* kc_out = out + (size_t)TT * HID;
  float* vc_out = kc_out + (size_t)NSLOTS * NKVH * DD;

  char* ws = (char*)d_ws;
  bf16*  hbf    = (bf16*)(ws);                         // 32 MB
  bf16*  wqkvbf = (bf16*)(ws + ((size_t)32  << 20));   // 48 MB
  bf16*  wobf   = (bf16*)(ws + ((size_t)80  << 20));   // 32 MB
  float* qkv    = (float*)(ws + ((size_t)112 << 20));  // 96 MB
  bf16*  qbf    = (bf16*)(ws + ((size_t)208 << 20));   // 32 MB
  bf16*  kbf    = (bf16*)(ws + ((size_t)240 << 20));   //  8 MB
  bf16*  vTb    = (bf16*)(ws + ((size_t)248 << 20));   //  8 MB
  bf16*  attnbf = (bf16*)(ws + ((size_t)256 << 20));   // 32 MB

  // 1) fp32 -> bf16 operands
  cvt_bf16_kernel<<<2048, 256, 0, stream>>>(hidden, hbf, (long)TT * HID);
  cvt_bf16_kernel<<<2048, 256, 0, stream>>>(wqkv, wqkvbf, (long)NQKV * HID);
  cvt_bf16_kernel<<<2048, 256, 0, stream>>>(wo, wobf, (long)HID * HID);

  // 2) QKV GEMM: [T,HID] x [NQKV,HID]^T -> [T,NQKV]
  gemm_bf16_tn<<<dim3(NQKV / 64, TT / 128), 128, 0, stream>>>(
      hbf, wqkvbf, qkv, TT, NQKV, HID);

  // 3) base cache copy (untouched slots must equal input zeros)
  copy_f32_kernel<<<2048, 256, 0, stream>>>(kcache_in, kc_out,
                                            (long)NSLOTS * NKVH * DD);
  copy_f32_kernel<<<2048, 256, 0, stream>>>(vcache_in, vc_out,
                                            (long)NSLOTS * NKVH * DD);

  // 4) RoPE + scatter + V-transpose
  rope_scatter_kernel<<<dim3(TT, NH), 64, 0, stream>>>(
      qkv, cosp, sinp, slots, qbf, kbf, vTb, kc_out, vc_out);

  // 5) causal flash attention
  flash_attn_kernel<<<dim3(SS / 16, 4 * NH), 32, 0, stream>>>(qbf, kbf, vTb,
                                                              attnbf);

  // 6) output projection: [T,HID] x [HID,HID]^T -> d_out
  gemm_bf16_tn<<<dim3(HID / 64, TT / 128), 128, 0, stream>>>(
      attnbf, wobf, out, TT, HID, HID);
}
